// RecurrentUnit_13520557048081
// MI455X (gfx1250) — matrix operand order repaired and checked by reading, not compiled, not run on hardware
//
#include <hip/hip_runtime.h>
#include <hip/hip_bf16.h>
#include <math.h>

// ---------------------------------------------------------------------------
// RecurrentUnit (point-cloud GRU) for MI455X / gfx1250, wave32 + WMMA f16.
//
// Roofline: ~23 GFLOP of 64-ch GEMMs, ~26MB working set (fits in 192MB L2)
// -> compute bound. All channel contractions use v_wmma_f32_16x16x32_f16
// with M-tile = the 16 KNN neighbors of one query, so the K-dim maxpool is
// an in-register reduction + one cross-half shuffle.
// ---------------------------------------------------------------------------

typedef __attribute__((ext_vector_type(16))) _Float16 v16h;
typedef __attribute__((ext_vector_type(8)))  _Float16 v8h;
typedef __attribute__((ext_vector_type(8)))  float    v8f;

#define Bb   2
#define Nn   8192
#define KNN  16
#define ROWS (Bb * Nn)      // 16384

// ---- WMMA helpers ----------------------------------------------------------

__device__ __forceinline__ v8f wmma_f16(v16h a, v16h b, v8f c) {
  // D = A(16x32 f16) * B(32x16 f16) + C(16x16 f32)
  return __builtin_amdgcn_wmma_f32_16x16x32_f16(
      /*neg_a=*/false, a, /*neg_b=*/false, b,
      /*c_mod=*/(short)0, c, /*reuse_a=*/false, /*reuse_b=*/false);
}

// A-matrix lane layout (16-bit A 16x32, ISA 7.12.2):
//   lanes 0-15 : M=lane,    elems 0..7 -> K=0..7,   elems 8..15 -> K=16..23
//   lanes 16-31: M=lane-16, elems 0..7 -> K=8..15,  elems 8..15 -> K=24..31
// Source is a row-major f16 [16][stride] LDS tile; both 8-half spans are
// 16-byte aligned (stride is a multiple of 8 halves).
__device__ __forceinline__ v16h load_a_tile(const _Float16* base, int lane,
                                            int kcbase, int stride) {
  int M  = lane & 15;
  int c0 = kcbase + ((lane & 16) ? 8 : 0);
  const _Float16* p = base + M * stride + c0;
  v8h lo = *(const v8h*)(p);
  v8h hi = *(const v8h*)(p + 16);
  v16h a;
#pragma unroll
  for (int e = 0; e < 8; ++e) { a[e] = lo[e]; a[e + 8] = hi[e]; }
  return a;
}

// B-matrix tiles are pre-swizzled by prep_weights into blob slots
// tile*512 + lane*16 + e, so each lane issues one 32-byte load.
__device__ __forceinline__ v16h load_b_tile(const _Float16* blob, int tile,
                                            int lane) {
  return *(const v16h*)(blob + (size_t)tile * 512 + lane * 16);
}

__device__ __forceinline__ float lrelu(float x) { return x > 0.f ? x : 0.1f * x; }
__device__ __forceinline__ float sigm(float x)  { return 1.f / (1.f + expf(-x)); }

// ---- K0: weight pre-swizzle into WMMA B lane layout (f16) ------------------
// B[k][col] = W[col][k];  lanes 0-15 hold K=0..7,16..23, lanes 16-31 K=8..15,24..31.
__global__ void prep_weights(const float* __restrict__ W,
                             _Float16* __restrict__ blob, int Cin) {
  int tile = blockIdx.x;            // kc*4 + nt
  int kc = tile >> 2, nt = tile & 3;
  int t = threadIdx.x;              // 0..511 == lane*16 + e
  int lane = t >> 4;
  int e    = t & 15;
  int o  = nt * 16 + (lane & 15);
  int kk = kc * 32 + ((lane & 16) ? 8 : 0) + ((e < 8) ? e : (e + 8));
  float v = (kk < Cin) ? W[o * Cin + kk] : 0.f;
  blob[(size_t)tile * 512 + t] = (_Float16)v;
}

// ---- K1: KNN (top-16 by squared distance, stable ascending) ----------------
__global__ void knn_kernel(const float* __restrict__ xyz,
                           int* __restrict__ knn_idx) {
  __shared__ float sxx[64], syy[64], szz[64];
  __shared__ float sdist[64][16];
  __shared__ int   sidx[64][16];

  int tid = threadIdx.x;            // 64 threads, one query each
  int q   = blockIdx.x * 64 + tid;
  int b   = q / Nn;
  int n   = q - b * Nn;
  const float* xb = xyz + (size_t)b * 3 * Nn;

  float qx = xb[n], qy = xb[Nn + n], qz = xb[2 * Nn + n];

#pragma unroll
  for (int j = 0; j < KNN; ++j) sdist[tid][j] = 3.4e38f;
  float dmax = 3.4e38f;

  for (int m0 = 0; m0 < Nn; m0 += 64) {
    __syncthreads();
    sxx[tid] = xb[m0 + tid];
    syy[tid] = xb[Nn + m0 + tid];
    szz[tid] = xb[2 * Nn + m0 + tid];
    if (m0 + 64 < Nn) __builtin_prefetch(xb + m0 + 64 + tid, 0, 0);
    __syncthreads();
    for (int jj = 0; jj < 64; ++jj) {
      float dx = sxx[jj] - qx, dy = syy[jj] - qy, dz = szz[jj] - qz;
      float d = dx * dx + dy * dy + dz * dz;
      if (d < dmax) {
        int pos = KNN - 1;
        while (pos > 0 && sdist[tid][pos - 1] > d) {
          sdist[tid][pos] = sdist[tid][pos - 1];
          sidx[tid][pos]  = sidx[tid][pos - 1];
          --pos;
        }
        sdist[tid][pos] = d;
        sidx[tid][pos]  = m0 + jj;
        dmax = sdist[tid][KNN - 1];
      }
    }
  }
#pragma unroll
  for (int j = 0; j < KNN; ++j) knn_idx[(size_t)q * KNN + j] = sidx[tid][j];
}

// ---- K2: fuse_r / fuse_z GEMMs: [ROWS x 64] x [64 x 64] --------------------
__global__ void fuse_kernel(const float* __restrict__ feats,
                            const _Float16* __restrict__ wb_fr,
                            const _Float16* __restrict__ wb_fz,
                            float* __restrict__ Fr, float* __restrict__ Fz) {
  __shared__ _Float16 sA[4][16 * 72];
  int lane = threadIdx.x & 31;
  int w    = threadIdx.x >> 5;
  int row0 = (blockIdx.x * 4 + w) * 16;
  int b    = row0 / Nn;
  int n0   = row0 - b * Nn;
  _Float16* aw = sA[w];

  // stage A: A[m][c] = feats[b][c][n0+m]  (each lane: two contiguous 64B spans)
#pragma unroll
  for (int t = 0; t < 2; ++t) {
    int c = lane + 32 * t;
    const float* src = feats + ((size_t)b * 64 + c) * Nn + n0;
#pragma unroll
    for (int m = 0; m < 16; ++m) aw[m * 72 + c] = (_Float16)src[m];
  }

  int o = (lane & 15);
  int mbase = (lane < 16) ? 0 : 8;
#pragma unroll
  for (int nt = 0; nt < 4; ++nt) {
    v8f ar = {0.f, 0.f, 0.f, 0.f, 0.f, 0.f, 0.f, 0.f};
    v8f az = {0.f, 0.f, 0.f, 0.f, 0.f, 0.f, 0.f, 0.f};
#pragma unroll
    for (int kc = 0; kc < 2; ++kc) {
      v16h a = load_a_tile(aw, lane, kc * 32, 72);
      ar = wmma_f16(a, load_b_tile(wb_fr, kc * 4 + nt, lane), ar);
      az = wmma_f16(a, load_b_tile(wb_fz, kc * 4 + nt, lane), az);
    }
    int oc = nt * 16 + o;
#pragma unroll
    for (int j = 0; j < 8; ++j) {
      size_t r = (size_t)(row0 + mbase + j) * 64 + oc;
      Fr[r] = ar[j];
      Fz[r] = az[j];
    }
  }
}

// ---- K3: per-query gated MLPs (conv0 x3, r-conv1, pexp, maxpools) ----------
__global__ void gru_main(const float* __restrict__ xyz,
                         const float* __restrict__ feats,
                         const float* __restrict__ cost,
                         const int* __restrict__ knn,
                         const float* __restrict__ Fr,
                         const float* __restrict__ Fz,
                         const float* __restrict__ b_r0,
                         const float* __restrict__ b_z0,
                         const float* __restrict__ b_h0,
                         const float* __restrict__ b_r1,
                         const _Float16* __restrict__ blob, // conv0|r1|ro
                         float* __restrict__ zpool,
                         float* __restrict__ hpool) {
  __shared__ _Float16 sA[4][16 * 104];  // new_points, padded 67 -> 96
  __shared__ _Float16 sR[4][16 * 72];   // r / gated staging

  int lane = threadIdx.x & 31;
  int w    = threadIdx.x >> 5;
  _Float16* aw = sA[w];
  _Float16* rw = sR[w];

  const _Float16* wb_c0 = blob;             // 36 tiles (3 gates x 3kc x 4nt)
  const _Float16* wb_r1 = blob + 18432;     // 8 tiles
  const _Float16* wb_ro = blob + 22528;     // 8 tiles

  // zero-pad once; per-query writes only touch cols 0..66
  for (int i = lane; i < 16 * 104; i += 32) aw[i] = (_Float16)0.f;

  int o16   = lane & 15;
  int mbase = (lane < 16) ? 0 : 8;

  for (int q = blockIdx.x * 4 + w; q < ROWS; q += gridDim.x * 4) {
    int b = q / Nn;
    int n = q - b * Nn;

    // ---- gather new_points[16 x 67] into LDS (f16, row-major stride 104)
    int k    = lane & 15;
    int idxk = knn[(size_t)q * KNN + k];
    int cb   = (lane >> 4) * 32;
    const float* csrc = cost + (size_t)b * 64 * Nn + idxk;
#pragma unroll
    for (int i = 0; i < 32; ++i) {
      int c = cb + i;
      aw[k * 104 + c] = (_Float16)csrc[(size_t)c * Nn];
    }
    if (lane < 16) {
      const float* xb = xyz + (size_t)b * 3 * Nn;
      aw[k * 104 + 64] = (_Float16)(xb[idxk] - xb[n]);
      aw[k * 104 + 65] = (_Float16)(xb[Nn + idxk] - xb[Nn + n]);
      aw[k * 104 + 66] = (_Float16)(xb[2 * Nn + idxk] - xb[2 * Nn + n]);
    }

    // ---- z gate: lrelu(conv0_z + b_z0 + Fz) -> maxpool over K
#pragma unroll
    for (int nt = 0; nt < 4; ++nt) {
      v8f acc = {0.f, 0.f, 0.f, 0.f, 0.f, 0.f, 0.f, 0.f};
#pragma unroll
      for (int kc = 0; kc < 3; ++kc)
        acc = wmma_f16(load_a_tile(aw, lane, kc * 32, 104),
                       load_b_tile(wb_c0, 12 + kc * 4 + nt, lane), acc);
      int o = nt * 16 + o16;
      float badd = b_z0[o] + Fz[(size_t)q * 64 + o];
      float mx = -3.4e38f;
#pragma unroll
      for (int j = 0; j < 8; ++j) mx = fmaxf(mx, lrelu(acc[j] + badd));
      mx = fmaxf(mx, __shfl_xor(mx, 16, 32));
      if (lane < 16) zpool[(size_t)q * 64 + o] = mx;
    }

    // ---- r gate: lrelu(conv0_r + b_r0 + Fr) -> LDS (A layout source)
#pragma unroll
    for (int nt = 0; nt < 4; ++nt) {
      v8f acc = {0.f, 0.f, 0.f, 0.f, 0.f, 0.f, 0.f, 0.f};
#pragma unroll
      for (int kc = 0; kc < 3; ++kc)
        acc = wmma_f16(load_a_tile(aw, lane, kc * 32, 104),
                       load_b_tile(wb_c0, 0 + kc * 4 + nt, lane), acc);
      int o = nt * 16 + o16;
      float badd = b_r0[o] + Fr[(size_t)q * 64 + o];
#pragma unroll
      for (int j = 0; j < 8; ++j)
        rw[(mbase + j) * 72 + o] = (_Float16)lrelu(acc[j] + badd);
    }

    // ---- r conv1 + sigmoid, then gated = r * f1  (back into sR)
    float r2[32];
#pragma unroll
    for (int nt = 0; nt < 4; ++nt) {
      v8f acc = {0.f, 0.f, 0.f, 0.f, 0.f, 0.f, 0.f, 0.f};
#pragma unroll
      for (int kc = 0; kc < 2; ++kc)
        acc = wmma_f16(load_a_tile(rw, lane, kc * 32, 72),
                       load_b_tile(wb_r1, kc * 4 + nt, lane), acc);
      int o = nt * 16 + o16;
#pragma unroll
      for (int j = 0; j < 8; ++j) r2[nt * 8 + j] = sigm(acc[j] + b_r1[o]);
    }
#pragma unroll
    for (int nt = 0; nt < 4; ++nt) {
      int o = nt * 16 + o16;
      float f1o = feats[((size_t)b * 64 + o) * Nn + n];
#pragma unroll
      for (int j = 0; j < 8; ++j)
        rw[(mbase + j) * 72 + o] = (_Float16)(r2[nt * 8 + j] * f1o);
    }

    // ---- h gate: lrelu(conv0_h + pexp + b_h0) -> maxpool over K
#pragma unroll
    for (int nt = 0; nt < 4; ++nt) {
      v8f acc = {0.f, 0.f, 0.f, 0.f, 0.f, 0.f, 0.f, 0.f};
#pragma unroll
      for (int kc = 0; kc < 2; ++kc)   // pexp = gated @ W_fuse_ro^T
        acc = wmma_f16(load_a_tile(rw, lane, kc * 32, 72),
                       load_b_tile(wb_ro, kc * 4 + nt, lane), acc);
#pragma unroll
      for (int kc = 0; kc < 3; ++kc)   // + conv0_h
        acc = wmma_f16(load_a_tile(aw, lane, kc * 32, 104),
                       load_b_tile(wb_c0, 24 + kc * 4 + nt, lane), acc);
      int o = nt * 16 + o16;
      float badd = b_h0[o];
      float mx = -3.4e38f;
#pragma unroll
      for (int j = 0; j < 8; ++j) mx = fmaxf(mx, lrelu(acc[j] + badd));
      mx = fmaxf(mx, __shfl_xor(mx, 16, 32));
      if (lane < 16) hpool[(size_t)q * 64 + o] = mx;
    }
  }
}

// ---- K4: z/h conv1 + GRU update -> feats_new [B,64,N], delta [row][64] -----
__global__ void tail_kernel(const float* __restrict__ feats,
                            const float* __restrict__ zpool,
                            const float* __restrict__ hpool,
                            const float* __restrict__ b_z1,
                            const float* __restrict__ b_h1,
                            const _Float16* __restrict__ wb_z1,
                            const _Float16* __restrict__ wb_h1,
                            float* __restrict__ out_feats,
                            float* __restrict__ delta) {
  __shared__ _Float16 sZ[4][16 * 72];
  __shared__ _Float16 sH[4][16 * 72];
  int lane = threadIdx.x & 31;
  int w    = threadIdx.x >> 5;
  int row0 = (blockIdx.x * 4 + w) * 16;
  int b    = row0 / Nn;
  int n0   = row0 - b * Nn;
  _Float16* zw = sZ[w];
  _Float16* hw = sH[w];

  {
    int m = lane & 15, half = lane >> 4;
    const float* zs = zpool + (size_t)(row0 + m) * 64 + half * 32;
    const float* hs = hpool + (size_t)(row0 + m) * 64 + half * 32;
#pragma unroll
    for (int i = 0; i < 32; ++i) {
      int c = half * 32 + i;
      zw[m * 72 + c] = (_Float16)zs[i];
      hw[m * 72 + c] = (_Float16)hs[i];
    }
  }

  int o16 = lane & 15;
  int mbase = (lane < 16) ? 0 : 8;
#pragma unroll
  for (int nt = 0; nt < 4; ++nt) {
    v8f az = {0.f, 0.f, 0.f, 0.f, 0.f, 0.f, 0.f, 0.f};
    v8f ah = {0.f, 0.f, 0.f, 0.f, 0.f, 0.f, 0.f, 0.f};
#pragma unroll
    for (int kc = 0; kc < 2; ++kc) {
      az = wmma_f16(load_a_tile(zw, lane, kc * 32, 72),
                    load_b_tile(wb_z1, kc * 4 + nt, lane), az);
      ah = wmma_f16(load_a_tile(hw, lane, kc * 32, 72),
                    load_b_tile(wb_h1, kc * 4 + nt, lane), ah);
    }
    int o = nt * 16 + o16;
    float bz = b_z1[o], bh = b_h1[o];
    const float* f1p = feats + ((size_t)b * 64 + o) * Nn + n0 + mbase;
    float* ofp = out_feats + ((size_t)b * 64 + o) * Nn + n0 + mbase;
#pragma unroll
    for (int j = 0; j < 8; ++j) {
      float zz = sigm(az[j] + bz);
      float hh = tanhf(ah[j] + bh);
      float f1 = f1p[j];
      float d  = zz * (hh - f1);
      ofp[j] = f1 + d;
      delta[(size_t)(row0 + mbase + j) * 64 + o] = d;
    }
  }
}

// ---- K5: fc 64->3, clip, + flow -> flow_out [B,3,N] ------------------------
__global__ void flow_kernel(const float* __restrict__ delta,
                            const float* __restrict__ W_fc,
                            const float* __restrict__ b_fc,
                            const float* __restrict__ flow,
                            float* __restrict__ out_flow) {
  int row = blockIdx.x * blockDim.x + threadIdx.x;
  if (row >= ROWS) return;
  int b = row / Nn;
  int n = row - b * Nn;
  const float* d = delta + (size_t)row * 64;
  float a0 = 0.f, a1 = 0.f, a2 = 0.f;
#pragma unroll
  for (int c = 0; c < 64; ++c) {
    float v = d[c];
    a0 += v * W_fc[c];
    a1 += v * W_fc[64 + c];
    a2 += v * W_fc[128 + c];
  }
  float f0 = fminf(fmaxf(a0 + b_fc[0], -200.f), 200.f);
  float f1 = fminf(fmaxf(a1 + b_fc[1], -200.f), 200.f);
  float f2 = fminf(fmaxf(a2 + b_fc[2], -200.f), 200.f);
  out_flow[((size_t)b * 3 + 0) * Nn + n] = f0 + flow[((size_t)b * 3 + 0) * Nn + n];
  out_flow[((size_t)b * 3 + 1) * Nn + n] = f1 + flow[((size_t)b * 3 + 1) * Nn + n];
  out_flow[((size_t)b * 3 + 2) * Nn + n] = f2 + flow[((size_t)b * 3 + 2) * Nn + n];
}

// ---------------------------------------------------------------------------
extern "C" void kernel_launch(void* const* d_in, const int* in_sizes, int n_in,
                              void* d_out, int out_size, void* d_ws,
                              size_t ws_size, hipStream_t stream) {
  const float* xyz    = (const float*)d_in[0];
  const float* feats  = (const float*)d_in[1];
  const float* cost   = (const float*)d_in[2];
  const float* flow   = (const float*)d_in[3];
  const float* W_r0   = (const float*)d_in[4];
  const float* b_r0   = (const float*)d_in[5];
  const float* W_r1   = (const float*)d_in[6];
  const float* b_r1   = (const float*)d_in[7];
  const float* W_z0   = (const float*)d_in[8];
  const float* b_z0   = (const float*)d_in[9];
  const float* W_z1   = (const float*)d_in[10];
  const float* b_z1   = (const float*)d_in[11];
  const float* W_h0   = (const float*)d_in[12];
  const float* b_h0   = (const float*)d_in[13];
  const float* W_h1   = (const float*)d_in[14];
  const float* b_h1   = (const float*)d_in[15];
  const float* W_fr   = (const float*)d_in[16];
  const float* W_fz   = (const float*)d_in[17];
  const float* W_fro  = (const float*)d_in[18];
  const float* W_fc   = (const float*)d_in[19];
  const float* b_fc   = (const float*)d_in[20];

  // ---- workspace carve-up (bytes) ----
  char* ws = (char*)d_ws;
  size_t off = 0;
  int*   knn   = (int*)(ws + off);   off += (size_t)ROWS * KNN * 4;   // 1 MB
  float* Fr    = (float*)(ws + off); off += (size_t)ROWS * 64 * 4;    // 4 MB
  float* Fz    = (float*)(ws + off); off += (size_t)ROWS * 64 * 4;    // 4 MB
  float* zpool = (float*)(ws + off); off += (size_t)ROWS * 64 * 4;    // 4 MB
  float* hpool = (float*)(ws + off); off += (size_t)ROWS * 64 * 4;    // 4 MB
  _Float16* blob = (_Float16*)(ws + off);                             // 86 KB
  float* delta = Fr;   // Fr is dead after gru_main; reuse for delta

  // blob half-offsets
  _Float16* wb_c0 = blob;           // 36 tiles: r(0..11) z(12..23) h(24..35)
  _Float16* wb_r1 = blob + 18432;   // 8 tiles
  _Float16* wb_ro = blob + 22528;
  _Float16* wb_fr = blob + 26624;
  _Float16* wb_fz = blob + 30720;
  _Float16* wb_z1 = blob + 34816;
  _Float16* wb_h1 = blob + 38912;

  // ---- K0: weight pre-swizzle (f16, WMMA B lane layout) ----
  prep_weights<<<12, 512, 0, stream>>>(W_r0, wb_c0,          67);
  prep_weights<<<12, 512, 0, stream>>>(W_z0, wb_c0 + 6144,   67);
  prep_weights<<<12, 512, 0, stream>>>(W_h0, wb_c0 + 12288,  67);
  prep_weights<<< 8, 512, 0, stream>>>(W_r1, wb_r1,          64);
  prep_weights<<< 8, 512, 0, stream>>>(W_fro, wb_ro,         64);
  prep_weights<<< 8, 512, 0, stream>>>(W_fr,  wb_fr,         64);
  prep_weights<<< 8, 512, 0, stream>>>(W_fz,  wb_fz,         64);
  prep_weights<<< 8, 512, 0, stream>>>(W_z1,  wb_z1,         64);
  prep_weights<<< 8, 512, 0, stream>>>(W_h1,  wb_h1,         64);

  // ---- K1: KNN ----
  knn_kernel<<<ROWS / 64, 64, 0, stream>>>(xyz, knn);

  // ---- K2: fuse_r / fuse_z GEMMs ----
  fuse_kernel<<<ROWS / 64, 128, 0, stream>>>(feats, wb_fr, wb_fz, Fr, Fz);

  // ---- K3: main gated-MLP WMMA kernel ----
  gru_main<<<128, 128, 0, stream>>>(xyz, feats, cost, knn, Fr, Fz,
                                    b_r0, b_z0, b_h0, b_r1, blob,
                                    zpool, hpool);

  // ---- K4: z/h conv1 + GRU update ----
  float* out_feats = (float*)d_out;
  float* out_flow  = (float*)d_out + (size_t)Bb * 64 * Nn;
  tail_kernel<<<ROWS / 64, 128, 0, stream>>>(feats, zpool, hpool, b_z1, b_h1,
                                             wb_z1, wb_h1, out_feats, delta);

  // ---- K5: fc + clip + flow residual ----
  flow_kernel<<<(ROWS + 255) / 256, 256, 0, stream>>>(delta, W_fc, b_fc, flow,
                                                      out_flow);
}
